// BertCounterFactCrossOpitionCompetitionTransformer_59648505806983
// MI455X (gfx1250) — matrix operand order repaired and checked by reading, not compiled, not run on hardware
//
#include <hip/hip_runtime.h>
#include <hip/hip_bf16.h>
#include <cstdint>

// ---------------------------------------------------------------------------
// BertCounterFactCrossOptionCompetitionTransformer for MI455X (gfx1250).
// B=32, L=512, D=768.
//
//  * f16 WMMA (v_wmma_f32_16x16x32_f16) for the GEMM-heavy stages.
//  * Weights pre-transposed to [N,K]: every wmma fragment load is a pair of
//    contiguous b128 loads per lane.
//  * Software-pipelined k-loops (double buffered fragments) so WMMAs overlap
//    the next iteration's VMEM instead of draining loadcnt to 0.
//  * proj: 16x64 tile/wave (4 accs, A reuse x4); scores: 16x32 tile/wave
//    (6 accs across 3 heads, A reuse x2).
//  * Gate-contraction: g2[b,m] = sum_l gate*attn, vec = g2 @ x  -- removes the
//    [B,L,D] attn@x GEMMs (~26 GFLOP + 100MB traffic) exactly.
// ---------------------------------------------------------------------------

typedef _Float16 v16h __attribute__((ext_vector_type(16)));
typedef _Float16 v8h  __attribute__((ext_vector_type(8)));
typedef float    v8f  __attribute__((ext_vector_type(8)));

constexpr int   Bn   = 32;
constexpr int   Ln   = 512;
constexpr int   Dn   = 768;
constexpr float NEGV = -9e15f;
constexpr float INVS = 0.036084391824351615f;   // 1/sqrt(768)
static const size_t NTOK = (size_t)Bn * Ln;     // 16384

// ---------------- WMMA fragment helpers (wave32, 16x16x32 f16) --------------

// A fragment: row-major source, K contiguous.  Lanes 0-15: rows 0-15 with
// K = {k0..k0+7, k0+16..k0+23}; lanes 16-31: same rows, K = {+8.., +24..}.
__device__ inline v16h load_a_frag(const _Float16* __restrict__ p, int ld,
                                   int row0, int k0) {
    int lane = threadIdx.x & 31;
    int r    = lane & 15;
    int hi   = lane >> 4;
    const _Float16* base = p + (size_t)(row0 + r) * ld + k0 + hi * 8;
    v8h lo = *reinterpret_cast<const v8h*>(base);
    v8h hh = *reinterpret_cast<const v8h*>(base + 16);
    v16h out;
#pragma unroll
    for (int j = 0; j < 8; ++j) { out[j] = lo[j]; out[8 + j] = hh[j]; }
    return out;
}

// B fragment from a K-contiguous source: element (k,n) stored at
// src[(n0+n)*ld + k]  (k_proj rows, or transposed weights Wt[n][k]).
__device__ inline v16h load_bT_frag(const _Float16* __restrict__ p, int ld,
                                    int n0, int k0) {
    int lane = threadIdx.x & 31;
    int n    = lane & 15;
    int hi   = lane >> 4;
    const _Float16* base = p + (size_t)(n0 + n) * ld + k0 + hi * 8;
    v8h lo = *reinterpret_cast<const v8h*>(base);
    v8h hh = *reinterpret_cast<const v8h*>(base + 16);
    v16h out;
#pragma unroll
    for (int j = 0; j < 8; ++j) { out[j] = lo[j]; out[8 + j] = hh[j]; }
    return out;
}

// C/D layout: VGPR v -> row v (lanes 0-15) / row v+8 (lanes 16-31), col = lane&15.
__device__ inline void store_c_f16(_Float16* __restrict__ p, int ld, int row0,
                                   int col0, v8f c, float bias) {
    int lane = threadIdx.x & 31;
    int n    = lane & 15;
    int hi   = lane >> 4;
#pragma unroll
    for (int v = 0; v < 8; ++v)
        p[(size_t)(row0 + v + hi * 8) * ld + col0 + n] = (_Float16)(c[v] + bias);
}

__device__ inline v8f wmma_f16(v16h a, v16h b, v8f c) {
    return __builtin_amdgcn_wmma_f32_16x16x32_f16(false, a, false, b,
                                                  (short)0, c, false, false);
}

// ---------------- casts / transpose -----------------------------------------

__global__ void cast_f32_to_f16(const float* __restrict__ src,
                                _Float16* __restrict__ dst, int n) {
    for (int i = blockIdx.x * blockDim.x + threadIdx.x; i < n;
         i += gridDim.x * blockDim.x)
        dst[i] = (_Float16)src[i];
}

struct Ptr6 { const float* p[6]; };

// Wt[z][n*Dn + k] = (f16) W[z][k*Dn + n]   (LDS-tiled 32x32 transpose)
__global__ void transpose_cast_kernel(Ptr6 W, _Float16* __restrict__ wt) {
    __shared__ float tile[32][33];
    int k0 = blockIdx.x * 32;
    int n0 = blockIdx.y * 32;
    int z  = blockIdx.z;
    const float* src = W.p[z];
    _Float16*    dst = wt + (size_t)z * Dn * Dn;
    int tx = threadIdx.x & 31;
    int ty = threadIdx.x >> 5;                       // block 256 = 32x8
#pragma unroll
    for (int i = ty; i < 32; i += 8)
        tile[i][tx] = src[(size_t)(k0 + i) * Dn + n0 + tx];
    __syncthreads();
#pragma unroll
    for (int i = ty; i < 32; i += 8)
        dst[(size_t)(n0 + i) * Dn + k0 + tx] = (_Float16)tile[tx][i];
}

// ---------------- masks + anomaly gate --------------------------------------

__global__ void mask_gate_kernel(const float* __restrict__ x,
                                 const int* __restrict__ ids,
                                 const int* __restrict__ padp,
                                 const int* __restrict__ sepp,
                                 const float* __restrict__ Wa,
                                 const float* __restrict__ ba,
                                 float* __restrict__ gate,
                                 unsigned char* __restrict__ fmask,
                                 unsigned char* __restrict__ omask) {
    int b = blockIdx.x, t = threadIdx.x;           // blockDim = 512
    __shared__ int   s_first;
    __shared__ int   si[Ln];
    __shared__ float sf[Ln];

    int pad = *padp, sep = *sepp;
    int tok = ids[b * Ln + t];
    int valid = (tok != pad) ? 1 : 0;
    bool is_sep = (tok == sep);

    if (t == 0) s_first = Ln;
    __syncthreads();
    if (is_sep) atomicMin(&s_first, t);
    si[t] = valid;
    __syncthreads();
    for (int s = Ln / 2; s > 0; s >>= 1) {
        if (t < s) si[t] += si[t + s];
        __syncthreads();
    }
    int vlen  = si[0];
    int first = s_first;
    int fb    = vlen / 2;
    if (fb < 1) fb = 1;
    if (fb > Ln - 2) fb = Ln - 2;
    int sep_pos = (first < Ln) ? first : fb;

    unsigned char fm = (t < sep_pos && valid) ? 1 : 0;
    unsigned char om = (t > sep_pos && valid) ? 1 : 0;
    fmask[b * Ln + t] = fm;
    omask[b * Ln + t] = om;

    const float* xr = x + ((size_t)b * Ln + t) * Dn;
    float acc = ba[0];
#pragma unroll 4
    for (int j = 0; j < Dn; ++j) acc += xr[j] * Wa[j];
    float a = fm ? acc : NEGV;

    sf[t] = a;
    __syncthreads();
    for (int s = Ln / 2; s > 0; s >>= 1) {
        if (t < s) sf[t] = fmaxf(sf[t], sf[t + s]);
        __syncthreads();
    }
    float amax = sf[0];
    __syncthreads();
    float e = __expf(a - amax) * (float)fm;
    sf[t] = e;
    __syncthreads();
    for (int s = Ln / 2; s > 0; s >>= 1) {
        if (t < s) sf[t] += sf[t + s];
        __syncthreads();
    }
    float ssum = fmaxf(sf[0], 1e-8f);
    gate[b * Ln + t] = e / ssum;
}

// ---------------- WMMA GEMM kernels -----------------------------------------

// qk[z] = xh @ Wt[z] + bias[z].  16x64 tile per wave, double-buffered k-loop.
__global__ void proj_kernel(const _Float16* __restrict__ xh,
                            const _Float16* __restrict__ wt,
                            Ptr6 bias,
                            _Float16* __restrict__ qk) {
    int wave  = threadIdx.x >> 5;                  // 4 waves / block
    int mtile = blockIdx.x * 4 + wave;             // 0..1023
    int row0  = mtile * 16;
    int col0  = blockIdx.y * 64;                   // 12 col-tiles of 64
    int z     = blockIdx.z;                        // 0..5
    const _Float16* Wz  = wt + (size_t)z * Dn * Dn;
    _Float16*       out = qk + (size_t)z * NTOK * Dn;

    v8f acc0 = {}, acc1 = {}, acc2 = {}, acc3 = {};
    int r = threadIdx.x & 15;

    // prologue: fragments for k0 = 0
    v16h a  = load_a_frag(xh, Dn, row0, 0);
    v16h b0 = load_bT_frag(Wz, Dn, col0,      0);
    v16h b1 = load_bT_frag(Wz, Dn, col0 + 16, 0);
    v16h b2 = load_bT_frag(Wz, Dn, col0 + 32, 0);
    v16h b3 = load_bT_frag(Wz, Dn, col0 + 48, 0);

#pragma unroll 2
    for (int k0 = 0; k0 < Dn - 32; k0 += 32) {
        __builtin_prefetch(xh + (size_t)(row0 + r) * Dn + k0 + 64, 0, 1);
        // issue next iteration's loads first ...
        v16h an  = load_a_frag(xh, Dn, row0, k0 + 32);
        v16h bn0 = load_bT_frag(Wz, Dn, col0,      k0 + 32);
        v16h bn1 = load_bT_frag(Wz, Dn, col0 + 16, k0 + 32);
        v16h bn2 = load_bT_frag(Wz, Dn, col0 + 32, k0 + 32);
        v16h bn3 = load_bT_frag(Wz, Dn, col0 + 48, k0 + 32);
        // ... then consume current fragments
        acc0 = wmma_f16(a, b0, acc0);
        acc1 = wmma_f16(a, b1, acc1);
        acc2 = wmma_f16(a, b2, acc2);
        acc3 = wmma_f16(a, b3, acc3);
        a = an; b0 = bn0; b1 = bn1; b2 = bn2; b3 = bn3;
    }
    // epilogue
    acc0 = wmma_f16(a, b0, acc0);
    acc1 = wmma_f16(a, b1, acc1);
    acc2 = wmma_f16(a, b2, acc2);
    acc3 = wmma_f16(a, b3, acc3);

    int n = threadIdx.x & 15;
    const float* bz = bias.p[z];
    store_c_f16(out, Dn, row0, col0,      acc0, bz[col0 + n]);
    store_c_f16(out, Dn, row0, col0 + 16, acc1, bz[col0 + 16 + n]);
    store_c_f16(out, Dn, row0, col0 + 32, acc2, bz[col0 + 32 + n]);
    store_c_f16(out, Dn, row0, col0 + 48, acc3, bz[col0 + 48 + n]);
}

// Per (b, 16-row tile, 32-col tile): sup/con/rep score tiles, masked epilogue.
//   ssup = where(pair, q_sup.k_sup*inv, NEG)
//   srep = where(pair, q_rep.k_rep*inv + tanh(q_con.k_con*inv), NEG)
__global__ void scores_kernel(const _Float16* __restrict__ qk,
                              const unsigned char* __restrict__ fmask,
                              const unsigned char* __restrict__ omask,
                              float* __restrict__ ssup,
                              float* __restrict__ srep) {
    int b    = blockIdx.z;
    int row0 = blockIdx.x * 16;
    int col0 = blockIdx.y * 32;
    size_t boff = (size_t)b * Ln * Dn;
    const _Float16* qs = qk + 0 * NTOK * Dn + boff;
    const _Float16* ks = qk + 1 * NTOK * Dn + boff;
    const _Float16* qc = qk + 2 * NTOK * Dn + boff;
    const _Float16* kc = qk + 3 * NTOK * Dn + boff;
    const _Float16* qr = qk + 4 * NTOK * Dn + boff;
    const _Float16* kr = qk + 5 * NTOK * Dn + boff;

    v8f as0 = {}, as1 = {}, ac0 = {}, ac1 = {}, ar0 = {}, ar1 = {};

    // prologue: fragments for k0 = 0
    v16h aS  = load_a_frag(qs, Dn, row0, 0);
    v16h bS0 = load_bT_frag(ks, Dn, col0,      0);
    v16h bS1 = load_bT_frag(ks, Dn, col0 + 16, 0);
    v16h aC  = load_a_frag(qc, Dn, row0, 0);
    v16h bC0 = load_bT_frag(kc, Dn, col0,      0);
    v16h bC1 = load_bT_frag(kc, Dn, col0 + 16, 0);
    v16h aR  = load_a_frag(qr, Dn, row0, 0);
    v16h bR0 = load_bT_frag(kr, Dn, col0,      0);
    v16h bR1 = load_bT_frag(kr, Dn, col0 + 16, 0);

    for (int k0 = 0; k0 < Dn - 32; k0 += 32) {
        int kn = k0 + 32;
        // next iteration's loads ...
        v16h aSn  = load_a_frag(qs, Dn, row0, kn);
        v16h bS0n = load_bT_frag(ks, Dn, col0,      kn);
        v16h bS1n = load_bT_frag(ks, Dn, col0 + 16, kn);
        v16h aCn  = load_a_frag(qc, Dn, row0, kn);
        v16h bC0n = load_bT_frag(kc, Dn, col0,      kn);
        v16h bC1n = load_bT_frag(kc, Dn, col0 + 16, kn);
        v16h aRn  = load_a_frag(qr, Dn, row0, kn);
        v16h bR0n = load_bT_frag(kr, Dn, col0,      kn);
        v16h bR1n = load_bT_frag(kr, Dn, col0 + 16, kn);
        // ... then the 6 WMMAs for the current k-step
        as0 = wmma_f16(aS, bS0, as0);
        as1 = wmma_f16(aS, bS1, as1);
        ac0 = wmma_f16(aC, bC0, ac0);
        ac1 = wmma_f16(aC, bC1, ac1);
        ar0 = wmma_f16(aR, bR0, ar0);
        ar1 = wmma_f16(aR, bR1, ar1);
        aS = aSn; bS0 = bS0n; bS1 = bS1n;
        aC = aCn; bC0 = bC0n; bC1 = bC1n;
        aR = aRn; bR0 = bR0n; bR1 = bR1n;
    }
    as0 = wmma_f16(aS, bS0, as0);
    as1 = wmma_f16(aS, bS1, as1);
    ac0 = wmma_f16(aC, bC0, ac0);
    ac1 = wmma_f16(aC, bC1, ac1);
    ar0 = wmma_f16(aR, bR0, ar0);
    ar1 = wmma_f16(aR, bR1, ar1);

    int lane = threadIdx.x & 31;
    int n    = lane & 15;
    int hi   = lane >> 4;
#pragma unroll
    for (int c = 0; c < 2; ++c) {
        int col = col0 + c * 16 + n;
        unsigned char om = omask[b * Ln + col];
        v8f vs = c ? as1 : as0;
        v8f vc = c ? ac1 : ac0;
        v8f vr = c ? ar1 : ar0;
#pragma unroll
        for (int v = 0; v < 8; ++v) {
            int row = row0 + v + hi * 8;
            unsigned char fm = fmask[b * Ln + row];
            bool pair = fm && om;
            float su = pair ? vs[v] * INVS : NEGV;
            float rp = pair ? (vr[v] * INVS + tanhf(vc[v] * INVS)) : NEGV;
            size_t idx = ((size_t)b * Ln + row) * Ln + col;
            ssup[idx] = su;
            srep[idx] = rp;
        }
    }
}

// ---------------- softmax / gated contractions ------------------------------

__global__ void softmax_kernel(float* __restrict__ s) {
    int row = blockIdx.x, b = blockIdx.y, t = threadIdx.x;
    float* p = s + ((size_t)b * Ln + row) * Ln;
    __shared__ float sh[256];
    float v0 = p[t], v1 = p[t + 256];
    sh[t] = fmaxf(v0, v1);
    __syncthreads();
    for (int st = 128; st > 0; st >>= 1) {
        if (t < st) sh[t] = fmaxf(sh[t], sh[t + st]);
        __syncthreads();
    }
    float mx = sh[0];
    __syncthreads();
    float e0 = __expf(v0 - mx), e1 = __expf(v1 - mx);
    sh[t] = e0 + e1;
    __syncthreads();
    for (int st = 128; st > 0; st >>= 1) {
        if (t < st) sh[t] += sh[t + st];
        __syncthreads();
    }
    float sum = sh[0];
    p[t]       = e0 / sum;
    p[t + 256] = e1 / sum;
}

// g2[b,m] = sum_l gate[b,l] * attn[b,l,m]   (column sums, coalesced)
__global__ void g2_kernel(const float* __restrict__ attn,
                          const float* __restrict__ gate,
                          float* __restrict__ g2) {
    int b = blockIdx.x, t = threadIdx.x;           // block=256
    __shared__ float sg[Ln];
    for (int i = t; i < Ln; i += 256) sg[i] = gate[b * Ln + i];
    __syncthreads();
#pragma unroll
    for (int i = 0; i < 2; ++i) {
        int m = t + i * 256;
        const float* col = attn + (size_t)b * Ln * Ln + m;
        float acc = 0.f;
        for (int l = 0; l < Ln; ++l) acc += sg[l] * col[(size_t)l * Ln];
        g2[b * Ln + m] = acc;
    }
}

// fused[b, :] = [ sum_l gate*x , sum_m g2_rep*x , sum_m g2_sup*x ]
__global__ void gatedsum_kernel(const float* __restrict__ x,
                                const float* __restrict__ gate,
                                const float* __restrict__ g2r,
                                const float* __restrict__ g2s,
                                float* __restrict__ fused) {
    int b = blockIdx.x, t = threadIdx.x;           // block=256
    __shared__ float sg[Ln], sr[Ln], ss[Ln];
    for (int i = t; i < Ln; i += 256) {
        sg[i] = gate[b * Ln + i];
        sr[i] = g2r[b * Ln + i];
        ss[i] = g2s[b * Ln + i];
    }
    __syncthreads();
#pragma unroll
    for (int i = 0; i < 3; ++i) {
        int d = t + i * 256;
        float a0 = 0.f, a1 = 0.f, a2 = 0.f;
        for (int l = 0; l < Ln; ++l) {
            float xv = x[((size_t)b * Ln + l) * Dn + d];
            a0 += sg[l] * xv;
            a1 += sr[l] * xv;
            a2 += ss[l] * xv;
        }
        float* fo = fused + (size_t)b * 3 * Dn;
        fo[d]          = a0;
        fo[Dn + d]     = a1;
        fo[2 * Dn + d] = a2;
    }
}

// ---------------- fusion MLP + LayerNorm ------------------------------------

__global__ void mlp1_kernel(const float* __restrict__ fused,
                            const float* __restrict__ W1,
                            const float* __restrict__ b1,
                            float* __restrict__ h) {
    int b = blockIdx.x;
    int n = blockIdx.y * 256 + threadIdx.x;
    __shared__ float sf[3 * Dn];
    for (int i = threadIdx.x; i < 3 * Dn; i += 256)
        sf[i] = fused[(size_t)b * 3 * Dn + i];
    __syncthreads();
    float acc = b1[n];
#pragma unroll 4
    for (int k = 0; k < 3 * Dn; ++k) acc += sf[k] * W1[(size_t)k * Dn + n];
    h[(size_t)b * Dn + n] = fmaxf(acc, 0.f);
}

__global__ void mlp2_kernel(const float* __restrict__ h,
                            const float* __restrict__ W2,
                            const float* __restrict__ b2,
                            float* __restrict__ o) {
    int b = blockIdx.x;
    int n = blockIdx.y * 256 + threadIdx.x;
    __shared__ float sh[Dn];
    for (int i = threadIdx.x; i < Dn; i += 256)
        sh[i] = h[(size_t)b * Dn + i];
    __syncthreads();
    float acc = b2[n];
#pragma unroll 4
    for (int k = 0; k < Dn; ++k) acc += sh[k] * W2[(size_t)k * Dn + n];
    o[(size_t)b * Dn + n] = acc;
}

__global__ void ln_kernel(const float* __restrict__ o,
                          const float* __restrict__ g,
                          const float* __restrict__ bb,
                          float* __restrict__ out) {
    int b = blockIdx.x, t = threadIdx.x;           // block=256, 3 elems/thread
    __shared__ float sred[256];
    float v[3];
    float s = 0.f;
#pragma unroll
    for (int i = 0; i < 3; ++i) {
        v[i] = o[(size_t)b * Dn + t + i * 256];
        s += v[i];
    }
    sred[t] = s;
    __syncthreads();
    for (int st = 128; st > 0; st >>= 1) {
        if (t < st) sred[t] += sred[t + st];
        __syncthreads();
    }
    float mu = sred[0] / Dn;
    __syncthreads();
    float q = 0.f;
#pragma unroll
    for (int i = 0; i < 3; ++i) { float d = v[i] - mu; q += d * d; }
    sred[t] = q;
    __syncthreads();
    for (int st = 128; st > 0; st >>= 1) {
        if (t < st) sred[t] += sred[t + st];
        __syncthreads();
    }
    float var  = sred[0] / Dn;
    float rinv = rsqrtf(var + 1e-5f);
#pragma unroll
    for (int i = 0; i < 3; ++i) {
        int d = t + i * 256;
        out[(size_t)b * Dn + d] = (v[i] - mu) * rinv * g[d] + bb[d];
    }
}

// ---------------------------------------------------------------------------

extern "C" void kernel_launch(void* const* d_in, const int* in_sizes, int n_in,
                              void* d_out, int out_size, void* d_ws, size_t ws_size,
                              hipStream_t stream) {
    const float* x    = (const float*)d_in[0];
    const int*   ids  = (const int*)d_in[1];
    const int*   padp = (const int*)d_in[2];
    const int*   sepp = (const int*)d_in[3];
    const float* Wa   = (const float*)d_in[4];
    const float* ba   = (const float*)d_in[5];
    Ptr6 Wqk, bias;
    for (int z = 0; z < 6; ++z) {
        Wqk.p[z]  = (const float*)d_in[6 + 2 * z];
        bias.p[z] = (const float*)d_in[7 + 2 * z];
    }
    const float* W1  = (const float*)d_in[18];
    const float* b1  = (const float*)d_in[19];
    const float* W2  = (const float*)d_in[20];
    const float* b2  = (const float*)d_in[21];
    const float* lng = (const float*)d_in[22];
    const float* lnb = (const float*)d_in[23];
    float* out = (float*)d_out;

    // ---- workspace layout (~252 MB) ----
    char* w = (char*)d_ws;
    auto alloc = [&](size_t bytes) {
        char* r = w;
        w += (bytes + 255) & ~(size_t)255;
        return r;
    };
    _Float16* xh    = (_Float16*)alloc(NTOK * Dn * 2);
    _Float16* wt    = (_Float16*)alloc((size_t)6 * Dn * Dn * 2);
    _Float16* qk    = (_Float16*)alloc((size_t)6 * NTOK * Dn * 2);
    float*    gate  = (float*)alloc(NTOK * 4);
    unsigned char* fmask = (unsigned char*)alloc(NTOK);
    unsigned char* omask = (unsigned char*)alloc(NTOK);
    float*    ssup  = (float*)alloc((size_t)Bn * Ln * Ln * 4);
    float*    srep  = (float*)alloc((size_t)Bn * Ln * Ln * 4);
    float*    g2s   = (float*)alloc(NTOK * 4);
    float*    g2r   = (float*)alloc(NTOK * 4);
    float*    fused = (float*)alloc((size_t)Bn * 3 * Dn * 4);
    float*    h1    = (float*)alloc((size_t)Bn * Dn * 4);
    float*    o2    = (float*)alloc((size_t)Bn * Dn * 4);

    // ---- 1) cast x to f16; transpose+cast weights to Wt[n][k] f16 ----
    cast_f32_to_f16<<<4096, 256, 0, stream>>>(x, xh, (int)(NTOK * Dn));
    {
        dim3 grid(Dn / 32, Dn / 32, 6);
        transpose_cast_kernel<<<grid, 256, 0, stream>>>(Wqk, wt);
    }

    // ---- 2) segment masks + anomaly gate ----
    mask_gate_kernel<<<Bn, Ln, 0, stream>>>(x, ids, padp, sepp, Wa, ba,
                                            gate, fmask, omask);

    // ---- 3) six Q/K projections (WMMA, 16x64 tiles, pipelined) ----
    {
        dim3 grid(256, Dn / 64, 6);          // 4 waves/block * 256 = 1024 m-tiles
        proj_kernel<<<grid, 128, 0, stream>>>(xh, wt, bias, qk);
    }

    // ---- 4) masked score tiles (WMMA, 3 heads fused, pipelined) ----
    {
        dim3 grid(Ln / 16, Ln / 32, Bn);
        scores_kernel<<<grid, 32, 0, stream>>>(qk, fmask, omask, ssup, srep);
    }

    // ---- 5) row softmax (sup, rep) ----
    {
        dim3 grid(Ln, Bn);
        softmax_kernel<<<grid, 256, 0, stream>>>(ssup);
        softmax_kernel<<<grid, 256, 0, stream>>>(srep);
    }

    // ---- 6) gate-contracted attention: g2 = gate^T @ attn ----
    g2_kernel<<<Bn, 256, 0, stream>>>(ssup, gate, g2s);
    g2_kernel<<<Bn, 256, 0, stream>>>(srep, gate, g2r);

    // ---- 7) gated sums over x -> fused [B, 3D] ----
    gatedsum_kernel<<<Bn, 256, 0, stream>>>(x, gate, g2r, g2s, fused);

    // ---- 8) fusion MLP + LayerNorm ----
    {
        dim3 grid(Bn, Dn / 256);
        mlp1_kernel<<<grid, 256, 0, stream>>>(fused, W1, b1, h1);
        mlp2_kernel<<<grid, 256, 0, stream>>>(h1, W2, b2, o2);
    }
    ln_kernel<<<Bn, 256, 0, stream>>>(o2, lng, lnb, out);
}
